// GIN_75797582840349
// MI455X (gfx1250) — matrix operand order repaired and checked by reading, not compiled, not run on hardware
//
#include <hip/hip_runtime.h>
#include <hip/hip_bf16.h>
#include <stdint.h>

#define IN_CH  128
#define HID_CH 256
#define OUT_CH 128

typedef __attribute__((ext_vector_type(16))) __bf16 v16bf;
typedef __attribute__((ext_vector_type(8)))  float  v8f;

__device__ __forceinline__ unsigned short f32_to_bf16_rn(float f) {
    union { float f; unsigned u; } c; c.f = f;
    unsigned u = c.u;
    u += 0x7FFFu + ((u >> 16) & 1u);   // round-to-nearest-even
    return (unsigned short)(u >> 16);
}
__device__ __forceinline__ float bf16_to_f32(unsigned short h) {
    union { unsigned u; float f; } c; c.u = ((unsigned)h) << 16;
    return c.f;
}

// ---------------------------------------------------------------------------
// Kernel 1: aggr = (1 + eps) * x   (self-loop term folded in before scatter)
// ---------------------------------------------------------------------------
__global__ void gin_init_aggr(const float* __restrict__ x,
                              const float* __restrict__ eps,
                              float* __restrict__ aggr, int n_elems) {
    const float s = 1.0f + eps[0];
    const float4* __restrict__ x4 = (const float4*)x;
    float4* __restrict__ a4 = (float4*)aggr;
    int n4 = n_elems >> 2;
    for (int i = blockIdx.x * blockDim.x + threadIdx.x; i < n4;
         i += gridDim.x * blockDim.x) {
        float4 v = x4[i];
        v.x *= s; v.y *= s; v.z *= s; v.w *= s;
        a4[i] = v;
    }
}

// ---------------------------------------------------------------------------
// Kernel 2: transpose weights to [N][K] and split fp32 -> bf16 hi + lo
//   W1 [128][256] -> w1t_* [256][128];  W2 [256][128] -> w2t_* [128][256]
// ---------------------------------------------------------------------------
__global__ void gin_prep_weights(const float* __restrict__ W1,
                                 const float* __restrict__ W2,
                                 unsigned short* __restrict__ w1t_hi,
                                 unsigned short* __restrict__ w1t_lo,
                                 unsigned short* __restrict__ w2t_hi,
                                 unsigned short* __restrict__ w2t_lo) {
    int i = blockIdx.x * blockDim.x + threadIdx.x;
    if (i < IN_CH * HID_CH) {
        int k = i / HID_CH, n = i % HID_CH;
        float v = W1[i];
        unsigned short hi = f32_to_bf16_rn(v);
        w1t_hi[n * IN_CH + k] = hi;
        w1t_lo[n * IN_CH + k] = f32_to_bf16_rn(v - bf16_to_f32(hi));
    }
    if (i < HID_CH * OUT_CH) {
        int k = i / OUT_CH, n = i % OUT_CH;
        float v = W2[i];
        unsigned short hi = f32_to_bf16_rn(v);
        w2t_hi[n * HID_CH + k] = hi;
        w2t_lo[n * HID_CH + k] = f32_to_bf16_rn(v - bf16_to_f32(hi));
    }
}

// ---------------------------------------------------------------------------
// Kernel 3: edge scatter-add.  One wave per edge; lane l moves channels
// 4l..4l+3 (coalesced 512B read of x[src], 4 f32 atomics into aggr[dst]).
// x and aggr are both L2-resident (51 MB each in 192 MB L2).
// ---------------------------------------------------------------------------
__global__ void gin_scatter(const float* __restrict__ x,
                            const long long* __restrict__ ei,
                            float* __restrict__ aggr, int n_edges) {
    long long gid = (long long)blockIdx.x * blockDim.x + threadIdx.x;
    long long e = gid >> 5;
    if (e >= n_edges) return;
    int lane = threadIdx.x & 31;
    int src = (int)ei[e];
    int dst = (int)ei[(long long)n_edges + e];
    const float4 v = *(const float4*)(x + (long long)src * IN_CH + lane * 4);
    float* d = aggr + (long long)dst * IN_CH + lane * 4;
    atomicAdd(d + 0, v.x);
    atomicAdd(d + 1, v.y);
    atomicAdd(d + 2, v.z);
    atomicAdd(d + 3, v.w);
}

// ---------------------------------------------------------------------------
// Kernel 4: fused MLP + log-softmax.  One wave handles 32 nodes (two 16-row
// M-tiles) so every B fragment fetched from L2 feeds 6 WMMAs instead of 3.
// Split-bf16 WMMA (A_hi*B_hi + A_hi*B_lo + A_lo*B_hi) ~= fp32 GEMM.
// ---------------------------------------------------------------------------

// Build an A-fragment (hi+lo) from a row-major fp32 tile in LDS.
// A k order used here (contiguous, same as B): lanes 0-15 row M=lane, K=base..base+15 contiguous;
// lanes 16-31 same rows, K shifted by +16.
__device__ __forceinline__ void load_a_frag(const float* __restrict__ row,
                                            v16bf& a_hi, v16bf& a_lo) {
    union { v16bf v; unsigned short u[16]; } uh, ul;
#pragma unroll
    for (int j = 0; j < 8; ++j) {
        float f0 = row[j];
        float f1 = row[8 + j];
        unsigned short h0 = f32_to_bf16_rn(f0);
        unsigned short h1 = f32_to_bf16_rn(f1);
        uh.u[j]     = h0;
        uh.u[8 + j] = h1;
        ul.u[j]     = f32_to_bf16_rn(f0 - bf16_to_f32(h0));
        ul.u[8 + j] = f32_to_bf16_rn(f1 - bf16_to_f32(h1));
    }
    a_hi = uh.v;
    a_lo = ul.v;
}

__global__ void __launch_bounds__(128)
gin_mlp(const float* __restrict__ aggr,
        const unsigned short* __restrict__ w1t_hi,
        const unsigned short* __restrict__ w1t_lo,
        const unsigned short* __restrict__ w2t_hi,
        const unsigned short* __restrict__ w2t_lo,
        const float* __restrict__ b1, const float* __restrict__ b2,
        float* __restrict__ out, int n_tiles) {
    extern __shared__ float smem[];
    const int lane = threadIdx.x & 31;
    const int wave = threadIdx.x >> 5;
    const int tile = blockIdx.x * 4 + wave;
    if (tile >= n_tiles) return;

    float* h0 = smem + wave * 12288;  // [32][128] fp32 (16 KB), reused for out
    float* h1 = h0 + 4096;            // [32][256] fp32 (32 KB)
    const long long node0 = (long long)tile * 32;

    const int half = lane >> 4;       // 0: lanes 0-15, 1: lanes 16-31
    const int l15  = lane & 15;

    // ---- stage input tile (32x128 fp32 = 16 KB) into LDS ----
    {
        const float4* __restrict__ src = (const float4*)(aggr + node0 * IN_CH);
        float4* __restrict__ dst = (float4*)h0;
#pragma unroll
        for (int i = 0; i < 32; ++i) dst[lane + i * 32] = src[lane + i * 32];
    }

    // ---- GEMM1: h1 = relu(h0 @ W1 + b1)  (K=128 -> 4 steps of 32) ----
    v8f acc1[2][16];
#pragma unroll
    for (int m = 0; m < 2; ++m)
#pragma unroll
        for (int nt = 0; nt < 16; ++nt)
#pragma unroll
            for (int r = 0; r < 8; ++r) acc1[m][nt][r] = 0.0f;

#pragma unroll
    for (int kt = 0; kt < 4; ++kt) {
        v16bf a_hi[2], a_lo[2];
        load_a_frag(h0 + l15 * IN_CH + kt * 32 + half * 16, a_hi[0], a_lo[0]);
        load_a_frag(h0 + (16 + l15) * IN_CH + kt * 32 + half * 16, a_hi[1], a_lo[1]);
#pragma unroll
        for (int nt = 0; nt < 16; ++nt) {
            const int n = nt * 16 + l15;
            // B k order used here (same as A): lane (n) 0-15 holds K 0-15, lanes 16-31 K 16-31,
            // contiguous in our [N][K] transposed weight -> one 32B load each.
            v16bf b_hi = *(const v16bf*)(w1t_hi + n * IN_CH + kt * 32 + half * 16);
            v16bf b_lo = *(const v16bf*)(w1t_lo + n * IN_CH + kt * 32 + half * 16);
#pragma unroll
            for (int m = 0; m < 2; ++m) {
                acc1[m][nt] = __builtin_amdgcn_wmma_f32_16x16x32_bf16(
                    false, a_hi[m], false, b_hi, (short)0, acc1[m][nt], false, false);
                acc1[m][nt] = __builtin_amdgcn_wmma_f32_16x16x32_bf16(
                    false, a_hi[m], false, b_lo, (short)0, acc1[m][nt], false, false);
                acc1[m][nt] = __builtin_amdgcn_wmma_f32_16x16x32_bf16(
                    false, a_lo[m], false, b_hi, (short)0, acc1[m][nt], false, false);
            }
        }
    }
    // bias + relu + park h1 in LDS (C layout: VGPR r, lane -> M=r+8*half, N=l15)
#pragma unroll
    for (int nt = 0; nt < 16; ++nt) {
        float bv = b1[nt * 16 + l15];
#pragma unroll
        for (int m = 0; m < 2; ++m)
#pragma unroll
            for (int r = 0; r < 8; ++r) {
                float v = fmaxf(acc1[m][nt][r] + bv, 0.0f);
                h1[(m * 16 + r + half * 8) * HID_CH + nt * 16 + l15] = v;
            }
    }

    // ---- GEMM2: o = relu(h1 @ W2 + b2)  (K=256 -> 8 steps of 32) ----
    v8f acc2[2][8];
#pragma unroll
    for (int m = 0; m < 2; ++m)
#pragma unroll
        for (int nt = 0; nt < 8; ++nt)
#pragma unroll
            for (int r = 0; r < 8; ++r) acc2[m][nt][r] = 0.0f;

#pragma unroll
    for (int kt = 0; kt < 8; ++kt) {
        v16bf a_hi[2], a_lo[2];
        load_a_frag(h1 + l15 * HID_CH + kt * 32 + half * 16, a_hi[0], a_lo[0]);
        load_a_frag(h1 + (16 + l15) * HID_CH + kt * 32 + half * 16, a_hi[1], a_lo[1]);
#pragma unroll
        for (int nt = 0; nt < 8; ++nt) {
            const int n = nt * 16 + l15;
            v16bf b_hi = *(const v16bf*)(w2t_hi + n * HID_CH + kt * 32 + half * 16);
            v16bf b_lo = *(const v16bf*)(w2t_lo + n * HID_CH + kt * 32 + half * 16);
#pragma unroll
            for (int m = 0; m < 2; ++m) {
                acc2[m][nt] = __builtin_amdgcn_wmma_f32_16x16x32_bf16(
                    false, a_hi[m], false, b_hi, (short)0, acc2[m][nt], false, false);
                acc2[m][nt] = __builtin_amdgcn_wmma_f32_16x16x32_bf16(
                    false, a_hi[m], false, b_lo, (short)0, acc2[m][nt], false, false);
                acc2[m][nt] = __builtin_amdgcn_wmma_f32_16x16x32_bf16(
                    false, a_lo[m], false, b_hi, (short)0, acc2[m][nt], false, false);
            }
        }
    }

    // ---- bias + relu, park 32x128 fp32 output in LDS (reuse h0 area) ----
    float* o = h0;
#pragma unroll
    for (int nt = 0; nt < 8; ++nt) {
        float bv = b2[nt * 16 + l15];
#pragma unroll
        for (int m = 0; m < 2; ++m)
#pragma unroll
            for (int r = 0; r < 8; ++r) {
                float v = fmaxf(acc2[m][nt][r] + bv, 0.0f);
                o[(m * 16 + r + half * 8) * OUT_CH + nt * 16 + l15] = v;
            }
    }

    // ---- row-wise log-softmax over 128 channels, wave32 shfl reductions ----
#pragma unroll 1
    for (int row = 0; row < 32; ++row) {
        float v0 = o[row * OUT_CH + lane];
        float v1 = o[row * OUT_CH + lane + 32];
        float v2 = o[row * OUT_CH + lane + 64];
        float v3 = o[row * OUT_CH + lane + 96];
        float m = fmaxf(fmaxf(v0, v1), fmaxf(v2, v3));
#pragma unroll
        for (int off = 16; off >= 1; off >>= 1)
            m = fmaxf(m, __shfl_xor(m, off, 32));
        float s = expf(v0 - m) + expf(v1 - m) + expf(v2 - m) + expf(v3 - m);
#pragma unroll
        for (int off = 16; off >= 1; off >>= 1)
            s += __shfl_xor(s, off, 32);
        float lse = m + logf(s);
        float* dst = out + (node0 + row) * OUT_CH;
        dst[lane]      = v0 - lse;
        dst[lane + 32] = v1 - lse;
        dst[lane + 64] = v2 - lse;
        dst[lane + 96] = v3 - lse;
    }
}

// ---------------------------------------------------------------------------
extern "C" void kernel_launch(void* const* d_in, const int* in_sizes, int n_in,
                              void* d_out, int out_size, void* d_ws, size_t ws_size,
                              hipStream_t stream) {
    const float*     x   = (const float*)d_in[0];
    const long long* ei  = (const long long*)d_in[1];   // int64 edge_index [2, E]
    const float*     W1  = (const float*)d_in[2];
    const float*     b1  = (const float*)d_in[3];
    const float*     W2  = (const float*)d_in[4];
    const float*     b2  = (const float*)d_in[5];
    const float*     eps = (const float*)d_in[6];

    const int n_node_elems = in_sizes[0];        // N * 128
    const int n_nodes      = n_node_elems / IN_CH;
    const int n_edges      = in_sizes[1] / 2;

    // workspace layout: aggr fp32 [N*128] | w1t_hi | w1t_lo | w2t_hi | w2t_lo
    float* aggr = (float*)d_ws;
    unsigned short* w1t_hi = (unsigned short*)(aggr + (size_t)n_nodes * IN_CH);
    unsigned short* w1t_lo = w1t_hi + IN_CH * HID_CH;
    unsigned short* w2t_hi = w1t_lo + IN_CH * HID_CH;
    unsigned short* w2t_lo = w2t_hi + HID_CH * OUT_CH;

    {
        int threads = 256;
        int blocks = (n_node_elems / 4 + threads - 1) / threads;
        gin_init_aggr<<<blocks, threads, 0, stream>>>(x, eps, aggr, n_node_elems);
    }
    {
        int threads = 256;
        int total = IN_CH * HID_CH;   // == HID_CH * OUT_CH
        int blocks = (total + threads - 1) / threads;
        gin_prep_weights<<<blocks, threads, 0, stream>>>(W1, W2, w1t_hi, w1t_lo,
                                                         w2t_hi, w2t_lo);
    }
    {
        long long tot = (long long)n_edges * 32;
        int threads = 256;
        int blocks = (int)((tot + threads - 1) / threads);
        gin_scatter<<<blocks, threads, 0, stream>>>(x, ei, aggr, n_edges);
    }
    {
        int n_tiles = n_nodes / 32;               // 3125 (exact)
        int blocks = (n_tiles + 3) / 4;           // 4 waves per block
        size_t lds = 4 * 12288 * sizeof(float);   // 192 KB
        gin_mlp<<<blocks, 128, lds, stream>>>(aggr, w1t_hi, w1t_lo, w2t_hi, w2t_lo,
                                              b1, b2, (float*)d_out, n_tiles);
    }
}
